// CasualMultiHeadAtten_70720931496732
// MI455X (gfx1250) — matrix operand order repaired and checked by reading, not compiled, not run on hardware
//
#include <hip/hip_runtime.h>

// ---------------------------------------------------------------------------
// Causal MHA for MI455X (gfx1250, wave32, WMMA).
// B=4, S=2048, D=1024, H=16, dk=64. ~206 GFLOP dense vs ~7us of HBM traffic
// at 23.3 TB/s => compute bound => bf16 WMMA (16x16x32) with f32 accumulate,
// flash-style streaming softmax (never materialize SxS scores).
// Round 2: GEMM wave tile widened to 32x64 (1.0 operand loads per WMMA),
// attention K-block staged through LDS with global_load_async_to_lds_b128
// (ASYNCcnt / s_wait_asynccnt), removing the 8x redundant K loads.
// ---------------------------------------------------------------------------

#ifndef USE_ASYNC_LDS
#define USE_ASYNC_LDS 1
#endif

typedef __bf16 bf16_t;
typedef __attribute__((ext_vector_type(16))) __bf16 v16bf;
typedef __attribute__((ext_vector_type(8)))  __bf16 v8bf;
typedef __attribute__((ext_vector_type(4)))  __bf16 v4bf;
typedef __attribute__((ext_vector_type(8)))  float   v8f;

__device__ __forceinline__ v8f wmma_bf16(v16bf a, v16bf b, v8f c) {
  return __builtin_amdgcn_wmma_f32_16x16x32_bf16(false, a, false, b,
                                                 (short)0, c, false, false);
}

__device__ __forceinline__ v16bf pack16(v8bf lo, v8bf hi) {
  v16bf r;
#pragma unroll
  for (int t = 0; t < 8; ++t) { r[t] = lo[t]; r[t + 8] = hi[t]; }
  return r;
}

// LDS aperture base has zero low 32 bits, and VDST of async-to-LDS ops is a
// wave-relative LDS byte offset, so low 32 bits of a generic LDS pointer are
// exactly the LDS address the hardware wants.
__device__ __forceinline__ unsigned lds_addr32(const void* p) {
  return (unsigned)(unsigned long long)(uintptr_t)p;
}

// Row reductions across the 16 lanes holding one WMMA C/D row (lanes 0-15 and
// 16-31 are independent halves). ds_swizzle xor-mode, and_mask=0x1f.
__device__ __forceinline__ float redmax16(float v) {
  v = fmaxf(v, __builtin_bit_cast(float, __builtin_amdgcn_ds_swizzle(__builtin_bit_cast(int, v), 0x041F)));
  v = fmaxf(v, __builtin_bit_cast(float, __builtin_amdgcn_ds_swizzle(__builtin_bit_cast(int, v), 0x081F)));
  v = fmaxf(v, __builtin_bit_cast(float, __builtin_amdgcn_ds_swizzle(__builtin_bit_cast(int, v), 0x101F)));
  v = fmaxf(v, __builtin_bit_cast(float, __builtin_amdgcn_ds_swizzle(__builtin_bit_cast(int, v), 0x201F)));
  return v;
}
__device__ __forceinline__ float redsum16(float v) {
  v += __builtin_bit_cast(float, __builtin_amdgcn_ds_swizzle(__builtin_bit_cast(int, v), 0x041F));
  v += __builtin_bit_cast(float, __builtin_amdgcn_ds_swizzle(__builtin_bit_cast(int, v), 0x081F));
  v += __builtin_bit_cast(float, __builtin_amdgcn_ds_swizzle(__builtin_bit_cast(int, v), 0x101F));
  v += __builtin_bit_cast(float, __builtin_amdgcn_ds_swizzle(__builtin_bit_cast(int, v), 0x201F));
  return v;
}

// ---------------------------------------------------------------------------
// f32 -> bf16 cast, 4 elements / thread.
// ---------------------------------------------------------------------------
__global__ void cast_f32_bf16_v4(const float* __restrict__ src,
                                 bf16_t* __restrict__ dst, int n4) {
  int i = blockIdx.x * blockDim.x + threadIdx.x;
  if (i >= n4) return;
  const float4 f = reinterpret_cast<const float4*>(src)[i];
  v4bf o;
  o[0] = (bf16_t)f.x; o[1] = (bf16_t)f.y; o[2] = (bf16_t)f.z; o[3] = (bf16_t)f.w;
  reinterpret_cast<v4bf*>(dst)[i] = o;
}

// ---------------------------------------------------------------------------
// NT GEMM: C[M,N] = A[M,K] * W[N,K]^T, bf16 in / f32 accumulate.
// WG = 256 threads = 8 waves; WG tile 128(M) x 128(N); wave tile 32x64
// (8 f32 accumulators). K-step 32 -> 8 v_wmma per step, 8 operand loads.
// W (2 MB bf16) is L2-resident (192 MB L2), x streams once.
//
// Operand layouts (contiguous k order used by this code, 16-bit, wave32):
//   A (16x32, row-in-lane):  lane m<16 holds row m, K 0..15;
//                            lane 16+m holds row m, K 16..31.
//   B (32x16, K across VGPRs): lane n<16 holds col n (== W row n), K 0..15
//                              contiguous; lane 16+n holds K 16..31.
// ---------------------------------------------------------------------------
template <bool OUT_BF16>
__global__ void __launch_bounds__(256)
gemm_nt_bf16(const bf16_t* __restrict__ A, const bf16_t* __restrict__ W,
             void* __restrict__ Cout, int M, int N, int K) {
  const int lane = threadIdx.x & 31;
  const int wave = threadIdx.x >> 5;
  const int lm   = lane & 15;
  const bool hi  = lane >= 16;

  const int wm = wave & 3;        // 0..3 -> M sub-tile (32 rows)
  const int wn = wave >> 2;       // 0..1 -> N sub-tile (64 cols)
  const int m0 = blockIdx.x * 128 + wm * 32;
  const int n0 = blockIdx.y * 128 + wn * 64;

  v8f acc[2][4] = {};

  for (int k0 = 0; k0 < K; k0 += 32) {
    v16bf a[2];
#pragma unroll
    for (int im = 0; im < 2; ++im) {
      const bf16_t* ap = A + (size_t)(m0 + im * 16 + lm) * K + k0 + (hi ? 16 : 0);
      a[im] = pack16(*(const v8bf*)ap, *(const v8bf*)(ap + 8));
    }
    v16bf b[4];
#pragma unroll
    for (int in = 0; in < 4; ++in) {
      const bf16_t* bp = W + (size_t)(n0 + in * 16 + lm) * K + k0 + (hi ? 16 : 0);
      b[in] = *(const v16bf*)bp;   // 32B, 32B-aligned (k0 multiple of 32)
    }
#pragma unroll
    for (int im = 0; im < 2; ++im)
#pragma unroll
      for (int in = 0; in < 4; ++in)
        acc[im][in] = wmma_bf16(a[im], b[in], acc[im][in]);
  }

  // C layout: VGPR r -> row (r | r+8 by lane half), col = lm.
#pragma unroll
  for (int im = 0; im < 2; ++im) {
#pragma unroll
    for (int in = 0; in < 4; ++in) {
#pragma unroll
      for (int r = 0; r < 8; ++r) {
        const int row = m0 + im * 16 + r + (hi ? 8 : 0);
        const int col = n0 + in * 16 + lm;
        const float v = acc[im][in][r];
        if (OUT_BF16)
          ((bf16_t*)Cout)[(size_t)row * N + col] = (bf16_t)v;
        else
          ((float*)Cout)[(size_t)row * N + col] = v;
      }
    }
  }
}

// ---------------------------------------------------------------------------
// Flash-style causal attention. Grid (S/128, H, B), 256 threads = 8 waves.
// Wave w owns query rows [qb + 16w, qb + 16w + 16). dk = 64.
// Per 32-key block:
//   K block (32x64 bf16, 4KB) staged to LDS with one async b128 per thread
//   (ASYNCcnt), V block transposed into LDS with regular loads + ds stores,
//   then: 4 WMMA scores, online softmax (swizzle row reductions), P bounced
//   through wave-private LDS (C-layout -> A-layout), 4 WMMA for PV.
// ---------------------------------------------------------------------------
__global__ void __launch_bounds__(256)
attn_causal(const bf16_t* __restrict__ Q, const bf16_t* __restrict__ Km,
            const bf16_t* __restrict__ Vm, bf16_t* __restrict__ Ctx) {
  constexpr int S = 2048, D = 1024, DK = 64;
  const int qt = blockIdx.x;       // query tile (128 rows)
  const int h  = blockIdx.y;
  const int b  = blockIdx.z;
  const int lane = threadIdx.x & 31;
  const int wave = threadIdx.x >> 5;
  const int lm   = lane & 15;
  const bool hi  = lane >= 16;

  const int qb = qt * 128;
  const int qw = qb + wave * 16;
  const size_t headOff = (size_t)h * DK;
  const size_t rowBase = (size_t)b * S;

  __shared__ bf16_t Kl[32][72];        // K block [key][d], pad 8 halves
  __shared__ bf16_t VT[64][40];        // V^T block [d][key], pad 8 halves
  __shared__ bf16_t Pl[8][16][40];     // per-wave P tile 16x32, pad 8

  // Q A-operands for this wave's 16 rows: two d-chunks, loaded once.
  v16bf qa[2];
#pragma unroll
  for (int t = 0; t < 2; ++t) {
    const bf16_t* qp = Q + (rowBase + qw + lm) * D + headOff + t * 32 + (hi ? 16 : 0);
    qa[t] = pack16(*(const v8bf*)qp, *(const v8bf*)(qp + 8));
  }

  v8f acc[4] = {};                     // 16 x 64 context (4 x 16-col chunks)
  float mrow[8], lrow[8];
#pragma unroll
  for (int r = 0; r < 8; ++r) { mrow[r] = -1e30f; lrow[r] = 0.f; }

  const int kend = qb + 128;           // causal bound for this WG
  for (int kb = 0; kb < kend; kb += 32) {
    __syncthreads();                   // previous block's LDS reads done

    // --- async-stage K block: 32x64 bf16 = 4KB, one b128 per thread --------
    {
      const int t   = threadIdx.x;
      const int key = t >> 3;                       // 0..31
      const int dc8 = (t & 7) * 8;                  // 0,8,...,56 (halves)
      const bf16_t* kg = Km + (rowBase + kb + key) * D + headOff + dc8;
#if USE_ASYNC_LDS
      asm volatile("global_load_async_to_lds_b128 %0, %1, off"
                   :: "v"(lds_addr32(&Kl[key][dc8])),
                      "v"((unsigned long long)(uintptr_t)kg)
                   : "memory");
#else
      *(v8bf*)&Kl[key][dc8] = *(const v8bf*)kg;
#endif
    }

    // --- cooperative V transpose into LDS: 2048 elems / 256 thr = 8 each ---
    {
      const int t   = threadIdx.x;
      const int key = t >> 3;                       // 0..31
      const int dch = (t & 7) * 8;                  // 0,8,...,56
      const v8bf vv = *(const v8bf*)(Vm + (rowBase + kb + key) * D + headOff + dch);
#pragma unroll
      for (int e = 0; e < 8; ++e) VT[dch + e][key] = vv[e];
    }
#if USE_ASYNC_LDS
    asm volatile("s_wait_asynccnt 0x0" ::: "memory");
#endif
    __syncthreads();                   // K + V blocks visible to all waves

    // --- scores S = Q K^T (16 x 32), f32, K operand from LDS ---------------
    v8f s[2] = {};
#pragma unroll
    for (int kt = 0; kt < 2; ++kt) {
#pragma unroll
      for (int t = 0; t < 2; ++t) {
        const bf16_t* kp = &Kl[kt * 16 + lm][t * 32 + (hi ? 16 : 0)];
        const v16bf bv = pack16(*(const v8bf*)kp, *(const v8bf*)(kp + 8));
        s[kt] = wmma_bf16(qa[t], bv, s[kt]);
      }
    }

    // --- scale, causal mask, online softmax update, P to LDS ---------------
#pragma unroll
    for (int r = 0; r < 8; ++r) {
      const int qrow = qw + r + (hi ? 8 : 0);
      float x0 = s[0][r] * 0.125f;                  // 1/sqrt(64)
      float x1 = s[1][r] * 0.125f;
      if (kb + lm      > qrow) x0 = -1e30f;
      if (kb + 16 + lm > qrow) x1 = -1e30f;
      const float mnew = fmaxf(mrow[r], redmax16(fmaxf(x0, x1)));
      const float corr = __expf(mrow[r] - mnew);
      const float p0 = __expf(x0 - mnew);
      const float p1 = __expf(x1 - mnew);
      lrow[r] = lrow[r] * corr + redsum16(p0 + p1);
      mrow[r] = mnew;
#pragma unroll
      for (int c = 0; c < 4; ++c) acc[c][r] *= corr;
      const int prow = r + (hi ? 8 : 0);
      Pl[wave][prow][lm]      = (bf16_t)p0;
      Pl[wave][prow][16 + lm] = (bf16_t)p1;
    }

    // --- reload P in A layout (wave-private LDS; DS is in-order per wave) --
    const bf16_t* pp = &Pl[wave][lm][hi ? 16 : 0];
    const v16bf pa = pack16(*(const v8bf*)pp, *(const v8bf*)(pp + 8));

    // --- context += P V : B[k,n] = V[key k][c*16+n] = VT[c*16+n][k] --------
#pragma unroll
    for (int c = 0; c < 4; ++c) {
      const bf16_t* vp = &VT[c * 16 + lm][hi ? 16 : 0];
      const v16bf bv = pack16(*(const v8bf*)vp, *(const v8bf*)(vp + 8));
      acc[c] = wmma_bf16(pa, bv, acc[c]);
    }
  }

  // --- normalize and write context (bf16, d = h*64 + c*16 + lm) ------------
#pragma unroll
  for (int r = 0; r < 8; ++r) {
    const float inv = 1.0f / lrow[r];
    const int row = qw + r + (hi ? 8 : 0);
    const size_t base = (rowBase + row) * D + headOff;
#pragma unroll
    for (int c = 0; c < 4; ++c)
      Ctx[base + c * 16 + lm] = (bf16_t)(acc[c][r] * inv);
  }
}

// ---------------------------------------------------------------------------
// Host launcher. Workspace layout (bf16): x | Wq Wk Wv Wo | Q K V | Ctx
// Total = 2*(5*B*S*D + 4*D*D) = 92,274,688 bytes.
// ---------------------------------------------------------------------------
extern "C" void kernel_launch(void* const* d_in, const int* in_sizes, int n_in,
                              void* d_out, int out_size, void* d_ws,
                              size_t ws_size, hipStream_t stream) {
  (void)in_sizes; (void)n_in; (void)out_size; (void)ws_size;
  constexpr int    Bn = 4, Sn = 2048, Dn = 1024, Hn = 16;
  constexpr size_t BSD = (size_t)Bn * Sn * Dn;   // 8,388,608
  constexpr size_t DD  = (size_t)Dn * Dn;        // 1,048,576
  constexpr int    Mrows = Bn * Sn;              // 8192

  const float* x  = (const float*)d_in[0];
  const float* Wq = (const float*)d_in[1];
  const float* Wk = (const float*)d_in[2];
  const float* Wv = (const float*)d_in[3];
  const float* Wo = (const float*)d_in[4];

  bf16_t* w   = (bf16_t*)d_ws;
  bf16_t* xb  = w; w += BSD;
  bf16_t* Wqb = w; w += DD;
  bf16_t* Wkb = w; w += DD;
  bf16_t* Wvb = w; w += DD;
  bf16_t* Wob = w; w += DD;
  bf16_t* Qb  = w; w += BSD;
  bf16_t* Kb  = w; w += BSD;
  bf16_t* Vb  = w; w += BSD;
  bf16_t* Cx  = w; w += BSD;

  const int NT = 256;
  cast_f32_bf16_v4<<<(int)(BSD / 4 + NT - 1) / NT, NT, 0, stream>>>(x,  xb,  (int)(BSD / 4));
  cast_f32_bf16_v4<<<(int)(DD  / 4 + NT - 1) / NT, NT, 0, stream>>>(Wq, Wqb, (int)(DD / 4));
  cast_f32_bf16_v4<<<(int)(DD  / 4 + NT - 1) / NT, NT, 0, stream>>>(Wk, Wkb, (int)(DD / 4));
  cast_f32_bf16_v4<<<(int)(DD  / 4 + NT - 1) / NT, NT, 0, stream>>>(Wv, Wvb, (int)(DD / 4));
  cast_f32_bf16_v4<<<(int)(DD  / 4 + NT - 1) / NT, NT, 0, stream>>>(Wo, Wob, (int)(DD / 4));

  const dim3 gg(Mrows / 128, Dn / 128);          // (64, 8)
  gemm_nt_bf16<true ><<<gg, 256, 0, stream>>>(xb, Wqb, Qb, Mrows, Dn, Dn);
  gemm_nt_bf16<true ><<<gg, 256, 0, stream>>>(xb, Wkb, Kb, Mrows, Dn, Dn);
  gemm_nt_bf16<true ><<<gg, 256, 0, stream>>>(xb, Wvb, Vb, Mrows, Dn, Dn);

  attn_causal<<<dim3(Sn / 128, Hn, Bn), 256, 0, stream>>>(Qb, Kb, Vb, Cx);

  gemm_nt_bf16<false><<<gg, 256, 0, stream>>>(Cx, Wob, d_out, Mrows, Dn, Dn);
}
